// Hybrid_Spatial_Multihead_Self_Attention_34557306864283
// MI455X (gfx1250) — matrix operand order repaired and checked by reading, not compiled, not run on hardware
//
#include <hip/hip_runtime.h>
#include <hip/hip_bf16.h>

typedef __bf16 bf16;
typedef __attribute__((ext_vector_type(16))) __bf16 v16bf;
typedef __attribute__((ext_vector_type(8)))  __bf16 v8bf;
typedef __attribute__((ext_vector_type(8)))  float  v8f;

static __device__ __forceinline__ v8f wmma_bf16(v16bf a, v16bf b, v8f c) {
  return __builtin_amdgcn_wmma_f32_16x16x32_bf16(
      false, a, false, b, (short)0, c, false, false);
}

static __device__ __forceinline__ v8f vzero8() {
  v8f z;
#pragma unroll
  for (int i = 0; i < 8; ++i) z[i] = 0.0f;
  return z;
}

// A fragment: halves K = {hi*8+0..7, 16+hi*8+0..7} -> two 16B LDS loads.
static __device__ __forceinline__ v16bf ld_afrag(const bf16* row, int hi) {
  v8bf lo = *(const v8bf*)(row + hi * 8);
  v8bf hb = *(const v8bf*)(row + 16 + hi * 8);
  return __builtin_shufflevector(lo, hb, 0, 1, 2, 3, 4, 5, 6, 7,
                                 8, 9, 10, 11, 12, 13, 14, 15);
}

// P readback: interleaved (p0,p1) pairs -> A-fragment half order.
static __device__ __forceinline__ v16bf deinterleave(v16bf p) {
  return __builtin_shufflevector(p, p, 0, 2, 4, 6, 8, 10, 12, 14,
                                 1, 3, 5, 7, 9, 11, 13, 15);
}

static __device__ __forceinline__ v8bf cvt8(float4 f0, float4 f1) {
  v8bf p;
  p[0] = (bf16)f0.x; p[1] = (bf16)f0.y; p[2] = (bf16)f0.z; p[3] = (bf16)f0.w;
  p[4] = (bf16)f1.x; p[5] = (bf16)f1.y; p[6] = (bf16)f1.z; p[7] = (bf16)f1.w;
  return p;
}

#define DIM   224
#define NTOK  131072
#define XS_ST 40   // LDS row stride (bf16): rows stay 16B-aligned

// ---------------------------------------------------------------------------
// Kernel 1: fused QKV projection.  grid (NTOK/128, 2, 3), block 256 (8 waves).
// ---------------------------------------------------------------------------
__global__ __launch_bounds__(256) void qkv_gemm(
    const float* __restrict__ x,
    const float* __restrict__ Wq, const float* __restrict__ Wk,
    const float* __restrict__ Wv,
    bf16* __restrict__ qb, bf16* __restrict__ kb, bf16* __restrict__ vb) {
  const int mat = blockIdx.z;
  const float* W = (mat == 0) ? Wq : (mat == 1) ? Wk : Wv;
  bf16* dst      = (mat == 0) ? qb : (mat == 1) ? kb : vb;
  const int row0 = blockIdx.x * 128;
  const int col0 = blockIdx.y * 112;

  __shared__ __align__(64) bf16 xs[128 * XS_ST];
  __shared__ __align__(64) bf16 wsw[7 * 32 * 16];  // B-fragment order

  const int tid  = threadIdx.x;
  const int lane = tid & 31;
  const int wave = tid >> 5;
  const int hi   = lane >> 4;
  const int ln   = lane & 15;

  v8f acc[7];
#pragma unroll
  for (int t = 0; t < 7; ++t) acc[t] = vzero8();

  for (int k0 = 0; k0 < DIM; k0 += 32) {
    __syncthreads();
    // A slab: 8 contiguous floats/thread -> 2x b128 global load + 1x b128 store
#pragma unroll
    for (int idx = tid; idx < 512; idx += 256) {
      int r = idx >> 2, q = idx & 3;
      const float4* src = (const float4*)&x[(size_t)(row0 + r) * DIM + k0 + q * 8];
      *(v8bf*)&xs[r * XS_ST + q * 8] = cvt8(src[0], src[1]);
    }
    // W slab swizzled to B-fragment layout; 4 consecutive K -> one 8B store
    for (int idx = tid; idx < 896; idx += 256) {
      int kg = idx / 112, c = idx % 112;
      int t = c >> 4, n = c & 15;
      int k = kg * 4, kh = (k >> 3) & 1, h = (k & 7) + ((k >> 4) << 3);
      const float* wp = &W[(size_t)(k0 + k) * DIM + col0 + c];
      union { bf16 h4[4]; unsigned long long u; } pk;
#pragma unroll
      for (int j = 0; j < 4; ++j) pk.h4[j] = (bf16)wp[(size_t)j * DIM];
      *(unsigned long long*)&wsw[(((t << 5) + (kh << 4) + n) << 4) + h] = pk.u;
    }
    if (k0 + 32 < DIM)
      __builtin_prefetch(&x[(size_t)(row0 + (tid >> 1)) * DIM + k0 + 32], 0, 1);
    __syncthreads();

    // preload A + all 7 B fragments, then issue WMMAs back-to-back
    v16bf a = ld_afrag(&xs[(wave * 16 + ln) * XS_ST], hi);
    v16bf bfr[7];
#pragma unroll
    for (int t = 0; t < 7; ++t)
      bfr[t] = *(const v16bf*)&wsw[(((t << 5) + lane) << 4)];
#pragma unroll
    for (int t = 0; t < 7; ++t) acc[t] = wmma_bf16(a, bfr[t], acc[t]);
  }

#pragma unroll
  for (int t = 0; t < 7; ++t)
#pragma unroll
    for (int i = 0; i < 8; ++i)
      dst[(size_t)(row0 + wave * 16 + i + 8 * hi) * DIM + col0 + t * 16 + ln] =
          (bf16)acc[t][i];
}

// ---------------------------------------------------------------------------
// Kernel 2/3: flash-style windowed / grid attention (shared).
// grid (outerN, 4 heads, 8 batch), block 128 (4 waves).
// ---------------------------------------------------------------------------
__global__ __launch_bounds__(128) void attn_kernel(
    const bf16* __restrict__ qg, const bf16* __restrict__ kg,
    const bf16* __restrict__ vg, const float* __restrict__ pos,
    const float* __restrict__ scale_p, bf16* __restrict__ og,
    int seq, int chanOff, int branch) {
  __shared__ __align__(64) bf16 qs[256 * 32];
  __shared__ __align__(64) bf16 ks[256 * 32];
  __shared__ __align__(64) bf16 vt[32 * 256];     // transposed: [d][j]
  __shared__ __align__(64) bf16 ps[4 * 16 * 32];  // per-wave (p0,p1) pairs

  const int outer = blockIdx.x;
  const int head  = blockIdx.y;
  const int bb    = blockIdx.z;
  const int tid   = threadIdx.x;
  const int lane  = tid & 31;
  const int wave  = tid >> 5;
  const int hi    = lane >> 4;
  const int ln    = lane & 15;
  const float scale = scale_p[0];

  auto grow = [&](int j) -> size_t {
    int wy, wx, iy, ix;
    if (branch == 0) { wy = outer >> 4; wx = outer & 15; iy = j >> 3;     ix = j & 7; }
    else             { wy = j >> 4;     wx = j & 15;     iy = outer >> 3; ix = outer & 7; }
    return (size_t)bb * 16384 + (size_t)((wy * 8 + iy) * 128 + wx * 8 + ix);
  };

  const int cbase = chanOff + head * 28;  // even -> u32 loads are 4B aligned
  for (int idx = tid; idx < seq * 16; idx += 128) {
    int j = idx >> 4, d = (idx & 15) * 2;
    float q0 = 0.f, q1 = 0.f, k0v = 0.f, k1v = 0.f, v0 = 0.f, v1 = 0.f;
    if (d < 28) {
      size_t base = grow(j) * DIM + cbase + d;
      union { unsigned u; bf16 h[2]; } uq, uk, uv;
      uq.u = *(const unsigned*)(qg + base);
      uk.u = *(const unsigned*)(kg + base);
      uv.u = *(const unsigned*)(vg + base);
      q0 = (float)uq.h[0]; q1 = (float)uq.h[1];
      k0v = (float)uk.h[0]; k1v = (float)uk.h[1];
      v0 = (float)uv.h[0]; v1 = (float)uv.h[1];
    }
    union { unsigned u; bf16 h[2]; } sq, sk;
    sq.h[0] = (bf16)(q0 * scale); sq.h[1] = (bf16)(q1 * scale);
    sk.h[0] = (bf16)k0v;          sk.h[1] = (bf16)k1v;
    *(unsigned*)&qs[j * 32 + d] = sq.u;
    *(unsigned*)&ks[j * 32 + d] = sk.u;
    vt[d * seq + j]       = (bf16)v0;
    vt[(d + 1) * seq + j] = (bf16)v1;
  }
  __syncthreads();

  bf16* pb = &ps[wave * 512];
  unsigned* pb32 = (unsigned*)pb;
  const int nrt = seq >> 4, nkp = seq >> 5;

  for (int rt = wave; rt < nrt; rt += 4) {
    v16bf aq = ld_afrag(&qs[(rt * 16 + ln) * 32], hi);
    v8f o0 = vzero8(), o1 = vzero8();
    float rm[8], rs[8];
#pragma unroll
    for (int i = 0; i < 8; ++i) { rm[i] = -3.0e38f; rs[i] = 0.f; }

    const float* posr = pos + ((size_t)head * seq + rt * 16 + 8 * hi) * seq;

    for (int kp = 0; kp < nkp; ++kp) {
      const int j0 = kp * 32;
      v16bf bk0 = ld_afrag(&ks[(j0 + ln) * 32], hi);
      v16bf bk1 = ld_afrag(&ks[(j0 + 16 + ln) * 32], hi);
      v8f s0 = wmma_bf16(aq, bk0, vzero8());
      v8f s1 = wmma_bf16(aq, bk1, vzero8());

#pragma unroll
      for (int i = 0; i < 8; ++i) {
        float p0v = s0[i] + posr[(size_t)i * seq + j0 + ln];
        float p1v = s1[i] + posr[(size_t)i * seq + j0 + 16 + ln];
        float mloc = fmaxf(p0v, p1v);
        mloc = fmaxf(mloc, __shfl_xor(mloc, 1, 32));
        mloc = fmaxf(mloc, __shfl_xor(mloc, 2, 32));
        mloc = fmaxf(mloc, __shfl_xor(mloc, 4, 32));
        mloc = fmaxf(mloc, __shfl_xor(mloc, 8, 32));
        float nm    = fmaxf(rm[i], mloc);
        float alpha = __expf(rm[i] - nm);
        float p0 = __expf(p0v - nm), p1 = __expf(p1v - nm);
        float sl = p0 + p1;
        sl += __shfl_xor(sl, 1, 32);
        sl += __shfl_xor(sl, 2, 32);
        sl += __shfl_xor(sl, 4, 32);
        sl += __shfl_xor(sl, 8, 32);
        rs[i] = rs[i] * alpha + sl;
        rm[i] = nm;
        o0[i] *= alpha; o1[i] *= alpha;
        union { bf16 h[2]; unsigned u; } pk;
        pk.h[0] = (bf16)p0; pk.h[1] = (bf16)p1;
        pb32[(i + 8 * hi) * 16 + ln] = pk.u;
      }

      v16bf ap = deinterleave(*(const v16bf*)&pb[ln * 32 + hi * 16]);
      v16bf bv0 = ld_afrag(&vt[ln * seq + j0], hi);
      v16bf bv1 = ld_afrag(&vt[(16 + ln) * seq + j0], hi);
      o0 = wmma_bf16(ap, bv0, o0);
      o1 = wmma_bf16(ap, bv1, o1);
    }

#pragma unroll
    for (int i = 0; i < 8; ++i) {
      float inv = (rs[i] > 0.f) ? (1.0f / rs[i]) : 0.f;
      size_t base = grow(rt * 16 + i + 8 * hi) * DIM + cbase;
      og[base + ln] = (bf16)(o0[i] * inv);
      if (ln < 12) og[base + 16 + ln] = (bf16)(o1[i] * inv);
    }
  }
}

// ---------------------------------------------------------------------------
// Kernel 4: output projection  out = attn(bf16) @ Wo + bo  -> f32
// ---------------------------------------------------------------------------
__global__ __launch_bounds__(256) void out_gemm(
    const bf16* __restrict__ a_in, const float* __restrict__ Wo,
    const float* __restrict__ bo, float* __restrict__ out) {
  const int row0 = blockIdx.x * 128;
  const int col0 = blockIdx.y * 112;

  __shared__ __align__(64) bf16 xs[128 * XS_ST];
  __shared__ __align__(64) bf16 wsw[7 * 32 * 16];

  const int tid = threadIdx.x, lane = tid & 31, wave = tid >> 5;
  const int hi = lane >> 4, ln = lane & 15;

  v8f acc[7];
#pragma unroll
  for (int t = 0; t < 7; ++t) acc[t] = vzero8();

  for (int k0 = 0; k0 < DIM; k0 += 32) {
    __syncthreads();
#pragma unroll
    for (int idx = tid; idx < 512; idx += 256) {
      int r = idx >> 2, q = idx & 3;
      *(v8bf*)&xs[r * XS_ST + q * 8] =
          *(const v8bf*)&a_in[(size_t)(row0 + r) * DIM + k0 + q * 8];
    }
    for (int idx = tid; idx < 896; idx += 256) {
      int kg = idx / 112, c = idx % 112;
      int t = c >> 4, n = c & 15;
      int k = kg * 4, kh = (k >> 3) & 1, h = (k & 7) + ((k >> 4) << 3);
      const float* wp = &Wo[(size_t)(k0 + k) * DIM + col0 + c];
      union { bf16 h4[4]; unsigned long long u; } pk;
#pragma unroll
      for (int j = 0; j < 4; ++j) pk.h4[j] = (bf16)wp[(size_t)j * DIM];
      *(unsigned long long*)&wsw[(((t << 5) + (kh << 4) + n) << 4) + h] = pk.u;
    }
    if (k0 + 32 < DIM)
      __builtin_prefetch(&a_in[(size_t)(row0 + (tid >> 1)) * DIM + k0 + 32], 0, 1);
    __syncthreads();

    v16bf a = ld_afrag(&xs[(wave * 16 + ln) * XS_ST], hi);
    v16bf bfr[7];
#pragma unroll
    for (int t = 0; t < 7; ++t)
      bfr[t] = *(const v16bf*)&wsw[(((t << 5) + lane) << 4)];
#pragma unroll
    for (int t = 0; t < 7; ++t) acc[t] = wmma_bf16(a, bfr[t], acc[t]);
  }

#pragma unroll
  for (int t = 0; t < 7; ++t) {
    float bias = bo[col0 + t * 16 + ln];
#pragma unroll
    for (int i = 0; i < 8; ++i)
      out[(size_t)(row0 + wave * 16 + i + 8 * hi) * DIM + col0 + t * 16 + ln] =
          acc[t][i] + bias;
  }
}

// ---------------------------------------------------------------------------
extern "C" void kernel_launch(void* const* d_in, const int* in_sizes, int n_in,
                              void* d_out, int out_size, void* d_ws, size_t ws_size,
                              hipStream_t stream) {
  const float* x     = (const float*)d_in[0];
  const float* Wq    = (const float*)d_in[1];
  const float* Wk    = (const float*)d_in[2];
  const float* Wv    = (const float*)d_in[3];
  const float* Wo    = (const float*)d_in[4];
  const float* bo    = (const float*)d_in[5];
  const float* scale = (const float*)d_in[6];
  const float* pos1  = (const float*)d_in[7];
  const float* pos2  = (const float*)d_in[8];
  float* out = (float*)d_out;

  const size_t MAT = (size_t)NTOK * DIM;
  bf16* qb = (bf16*)d_ws;
  bf16* kb = qb + MAT;
  bf16* vb = kb + MAT;
  bf16* ab = vb + MAT;

  qkv_gemm<<<dim3(NTOK / 128, 2, 3), 256, 0, stream>>>(x, Wq, Wk, Wv, qb, kb, vb);

  attn_kernel<<<dim3(256, 4, 8), 128, 0, stream>>>(qb, kb, vb, pos1, scale, ab,
                                                   64, 0, 0);
  attn_kernel<<<dim3(64, 4, 8), 128, 0, stream>>>(qb, kb, vb, pos2, scale, ab,
                                                  256, 112, 1);

  out_gemm<<<dim3(NTOK / 128, 2), 256, 0, stream>>>(ab, Wo, bo, out);
}
